// ScaledDotProductAttention_75771813036480
// MI455X (gfx1250) — matrix operand order repaired and checked by reading, not compiled, not run on hardware
//
#include <hip/hip_runtime.h>

typedef __bf16 bf16_t;
typedef __attribute__((ext_vector_type(16))) __bf16 v16bf;
typedef __attribute__((ext_vector_type(8)))  __bf16 v8bf;
typedef __attribute__((ext_vector_type(8)))  float  v8f;
typedef __attribute__((ext_vector_type(4)))  float  f32x4;
typedef __attribute__((ext_vector_type(2)))  float  f32x2;
typedef __attribute__((__vector_size__(4 * sizeof(int)))) int i32x4;

constexpr int B_ = 2, H_ = 16, S_ = 2048, D_ = 64;
constexpr int ROWS   = 16;          // Q rows per workgroup
constexpr int SLDS   = 2052;        // padded LDS row stride (floats): row*2052 %64 = 4*row -> b128 conflict-free
constexpr int NWAVES = 16;          // 512-thread blocks
constexpr int KPW    = S_ / NWAVES; // 128 keys per wave for score@V

// ---- optional CDNA5 async global->LDS copy path (guarded; falls back to VGPR copy) ----
#if defined(__has_builtin)
#if __has_builtin(__builtin_amdgcn_global_load_async_to_lds_b128) && \
    __has_builtin(__builtin_amdgcn_s_wait_asynccnt)
#define HAVE_ASYNC_CP 1
#endif
#endif
#ifndef HAVE_ASYNC_CP
#define HAVE_ASYNC_CP 0
#endif

// ---------------- prep 1: K -> bf16 copy (same [bh][n][d] layout) ----------------
__global__ __launch_bounds__(256)
void prep_kb(const float* __restrict__ k, bf16_t* __restrict__ kb)
{
  const size_t base = ((size_t)blockIdx.x * 256 + threadIdx.x) * 8;
  f32x4 a = *(const f32x4*)(k + base);
  f32x4 b = *(const f32x4*)(k + base + 4);
  v8bf o;
#pragma unroll
  for (int j = 0; j < 4; ++j) { o[j] = (bf16_t)a[j]; o[4 + j] = (bf16_t)b[j]; }
  *(v8bf*)(kb + base) = o;
}

// ---------------- prep 2: V -> bf16 transpose: vt[bh][d][kk] = V[bh][kk][d] ----------------
__global__ __launch_bounds__(256)
void prep_vt(const float* __restrict__ v, bf16_t* __restrict__ vt)
{
  constexpr int TK = 256;                    // keys per tile
  __shared__ bf16_t tile[TK * 68];           // padded stride
  const int tid = threadIdx.x;
  const int bh  = blockIdx.x >> 3;           // 32 (b,h)
  const int k0  = (blockIdx.x & 7) * TK;     // 8 key-chunks

  const float* src = v + ((size_t)bh * S_ + k0) * D_;
#pragma unroll
  for (int i = 0; i < (TK * D_) / (256 * 4); ++i) {      // 16 iters, coalesced f32x4
    const int lin = (i * 256 + tid) * 4;
    const int kk = lin >> 6, d = lin & 63;
    f32x4 x = *(const f32x4*)(src + lin);
    bf16_t* t = &tile[kk * 68 + d];
    t[0] = (bf16_t)x[0]; t[1] = (bf16_t)x[1]; t[2] = (bf16_t)x[2]; t[3] = (bf16_t)x[3];
  }
  __syncthreads();
  bf16_t* dst = vt + (size_t)bh * D_ * S_ + k0;
#pragma unroll
  for (int i = 0; i < (TK * D_) / (256 * 8); ++i) {      // 8 iters, coalesced 16B stores
    const int lin = (i * 256 + tid) * 8;
    const int d = lin >> 8, kk = lin & 255;
    v8bf o;
#pragma unroll
    for (int j = 0; j < 8; ++j) o[j] = tile[(kk + j) * 68 + d];
    *(v8bf*)(dst + (size_t)d * S_ + kk) = o;
  }
}

// ---------------- fused attention ----------------
__global__ __launch_bounds__(512)
void attn_fused(const float* __restrict__ q,
                const bf16_t* __restrict__ kmat,      // bf16 [bh][n][d]
                const bf16_t* __restrict__ vt,        // bf16 [bh][d][k]
                const unsigned char* __restrict__ mask,
                float* __restrict__ outp,
                float* __restrict__ attenp)
{
  __shared__ float score[ROWS * SLDS];                     // masked, scaled scores (fp32)
  __shared__ __align__(16) unsigned char mlds[ROWS * S_];  // mask block (async-copied)
  __shared__ float outred[ROWS * D_];
  __shared__ float rowmax[ROWS];
  __shared__ float rowinv[ROWS];

  const int tid  = threadIdx.x;
  const int w    = tid >> 5;                  // 0..15
  const int lane = tid & 31;
  const int l15  = lane & 15;
  const int hi   = lane >> 4;

  const int gid  = blockIdx.x;
  const int mblk = gid & (S_ / ROWS - 1);
  const int bh   = gid / (S_ / ROWS);
  const int b    = bh / H_;
  const int m0   = mblk * ROWS;

  const float*  qb  = q    + (size_t)bh * S_ * D_;
  const bf16_t* kbB = kmat + (size_t)bh * S_ * D_;
  const bf16_t* vtB = vt   + (size_t)bh * D_ * S_;

  // ---- async copy of the contiguous 32KB mask block into LDS ----
  {
    const unsigned char* msrc = mask + (size_t)b * S_ * S_ + (size_t)m0 * S_;
#pragma unroll
    for (int i = 0; i < (ROWS * S_) / (512 * 16); ++i) {   // 4 x 16B per thread
      const int off = (i * 512 + tid) * 16;
#if HAVE_ASYNC_CP
      __builtin_amdgcn_global_load_async_to_lds_b128(
          (__attribute__((address_space(1))) i32x4*)(msrc + off),
          (__attribute__((address_space(3))) i32x4*)(mlds + off), 0, 0);
#else
      *(f32x4*)(mlds + off) = *(const f32x4*)(msrc + off);
#endif
    }
  }

  // zero cross-wave reduction buffer
  outred[tid * 2] = 0.f;
  outred[tid * 2 + 1] = 0.f;

  // ---- Q A-fragments (hoisted; contiguous K per lane-half A layout) ----
  // 1/sqrt(64) scale folded into Q here (q*0.125 is exact in fp32), so the
  // phase-1 WMMA result is already the scaled score -> no per-tile v_mul.
  v16bf qa0, qa1;
  {
    const float* qp = qb + (size_t)(m0 + l15) * D_;
    const int b8 = hi << 4;
#pragma unroll
    for (int i = 0; i < 8; ++i) {
      qa0[i]     = (bf16_t)(qp[b8 + i] * 0.125f);
      qa0[8 + i] = (bf16_t)(qp[8 + b8 + i] * 0.125f);
      qa1[i]     = (bf16_t)(qp[32 + b8 + i] * 0.125f);
      qa1[8 + i] = (bf16_t)(qp[40 + b8 + i] * 0.125f);
    }
  }

#if HAVE_ASYNC_CP
  __builtin_amdgcn_s_wait_asynccnt(0);
#endif
  __syncthreads();   // mask block visible to all waves; outred zeroed

  // ---------------- Phase 1: score = mask ? -1e9 : (Q K^T)/8 -> LDS ----------------
  for (int j = 0; j < S_ / 16 / NWAVES; ++j) {        // 8 tiles per wave
    const int nt  = w + NWAVES * j;
    const int col = nt * 16 + l15;
    const bf16_t* kp = kbB + (size_t)col * D_ + (hi << 4);  // B layout: contiguous K per lane-half
    if (j + 1 < S_ / 16 / NWAVES)
      __builtin_prefetch(kbB + (size_t)(col + NWAVES * 16) * D_, 0, 0);
    v16bf kf0 = *(const v16bf*)(kp);        // dims hi*16 + 0..15   (32B aligned bf16 load)
    v16bf kf1 = *(const v16bf*)(kp + 32);   // dims 32 + hi*16 + 0..15
    v8f c = {};
    c = __builtin_amdgcn_wmma_f32_16x16x32_bf16(false, qa0, false, kf0, (short)0, c, false, false);
    c = __builtin_amdgcn_wmma_f32_16x16x32_bf16(false, qa1, false, kf1, (short)0, c, false, false);
#pragma unroll
    for (int r = 0; r < 8; ++r) {
      const int row = r + (hi << 3);
      const unsigned char mk = mlds[row * S_ + col];
      score[row * SLDS + col] = mk ? -1e9f : c[r];
    }
  }

  __syncthreads();

  // ---------------- Softmax stats: one row per wave ----------------
  {
    const int row = w;
    const float* rp = &score[row * SLDS];
    float m = -3.4e38f;
#pragma unroll 4
    for (int i = lane; i < S_; i += 32) m = fmaxf(m, rp[i]);
#pragma unroll
    for (int off = 16; off >= 1; off >>= 1) m = fmaxf(m, __shfl_xor(m, off, 32));
    float s = 0.f;
#pragma unroll 4
    for (int i = lane; i < S_; i += 32) s += __expf(rp[i] - m);
#pragma unroll
    for (int off = 16; off >= 1; off >>= 1) s += __shfl_xor(s, off, 32);
    if (lane == 0) { rowmax[row] = m; rowinv[row] = 1.f / s; }
  }

  // ---------------- Phase 2: out += score[:, kw0:kw0+128] @ V (per-wave K split) ----------------
  {
    v8f acc[4] = {v8f{}, v8f{}, v8f{}, v8f{}};
    const int kw0 = w * KPW;
#pragma unroll
    for (int ks = 0; ks < KPW; ks += 32) {
      const int k0 = kw0 + ks;
      v16bf a;                                        // score fp32 -> bf16, contiguous A layout
      const float* rp = &score[l15 * SLDS + k0 + (hi << 4)];
#pragma unroll
      for (int i = 0; i < 8; ++i) {
        a[i]     = (bf16_t)rp[i];
        a[8 + i] = (bf16_t)rp[8 + i];
      }
#pragma unroll
      for (int t = 0; t < 4; ++t) {                   // 4 d-tiles; B frag = contiguous bf16 from Vt
        const v16bf vf = *(const v16bf*)(vtB + (size_t)(t * 16 + l15) * S_ + k0 + (hi << 4));
        acc[t] = __builtin_amdgcn_wmma_f32_16x16x32_bf16(false, a, false, vf,
                                                         (short)0, acc[t], false, false);
      }
    }
#pragma unroll
    for (int t = 0; t < 4; ++t)
#pragma unroll
      for (int r = 0; r < 8; ++r)
        atomicAdd(&outred[(r + (hi << 3)) * D_ + t * 16 + l15], acc[t][r]);   // ds_add_f32
  }

  __syncthreads();

  // ---------------- Final: stream atten (NT stores) + contiguous out block ----------------
  float* ab = attenp + ((size_t)bh * S_ + m0) * S_;   // 16 contiguous rows -> 128KB linear
  float* ob = outp   + ((size_t)bh * S_ + m0) * D_;

#pragma unroll 2
  for (int it = 0; it < (ROWS * S_) / (512 * 4); ++it) {   // 16 iterations of float4
    const int idx = (it * 512 + tid) * 4;
    const int row = idx >> 11;
    const float m = rowmax[row], inv = rowinv[row];
    const float* sp = &score[row * SLDS + (idx & (S_ - 1))];
    f32x4 p;
#pragma unroll
    for (int e = 0; e < 4; ++e) p[e] = __expf(sp[e] - m) * inv;
    __builtin_nontemporal_store(p, (f32x4*)(ab + idx));
  }

  {
    f32x2 o = *(f32x2*)&outred[tid * 2];
    __builtin_nontemporal_store(o, (f32x2*)(ob + tid * 2));
  }
}

extern "C" void kernel_launch(void* const* d_in, const int* in_sizes, int n_in,
                              void* d_out, int out_size, void* d_ws, size_t ws_size,
                              hipStream_t stream) {
  const float* q = (const float*)d_in[0];
  const float* k = (const float*)d_in[1];
  const float* v = (const float*)d_in[2];
  const unsigned char* mask = (const unsigned char*)d_in[3];  // numpy bool = 1 byte

  float* outp  = (float*)d_out;
  float* atten = (float*)d_out + (size_t)B_ * H_ * S_ * D_;

  // workspace: 8MB bf16 K copy + 8MB bf16 V^T (ws_size is expected to cover 16MB)
  bf16_t* kbuf  = (bf16_t*)d_ws;
  bf16_t* vtbuf = kbuf + (size_t)B_ * H_ * S_ * D_;

  prep_kb<<<dim3((B_ * H_ * S_ * D_) / (256 * 8)), 256, 0, stream>>>(k, kbuf);
  prep_vt<<<dim3(B_ * H_ * (S_ / 256)), 256, 0, stream>>>(v, vtbuf);
  attn_fused<<<dim3(B_ * H_ * (S_ / ROWS)), 512, 0, stream>>>(q, kbuf, vtbuf, mask, outp, atten);
}
